// DAWNBlock_21157008900530
// MI455X (gfx1250) — matrix operand order repaired and checked by reading, not compiled, not run on hardware
//
#include <hip/hip_runtime.h>
#include <math.h>

#define DEV __device__ __forceinline__

// ---------------------------------------------------------------------------
// Problem constants (B=4, S=1024, D=1024, H=16, DH=64, R=128, NC=NE=64,
// NK=4096, KR=128, TOPK=8)
// ---------------------------------------------------------------------------
#define BB 4
#define SS 1024
#define DD 1024
#define HH 16
#define DH 64
#define RR 128
#define NEXP 64
#define NK 4096
#define TOPK 8

typedef unsigned short u16;
typedef __attribute__((ext_vector_type(16))) __bf16 bf16x16;
typedef __attribute__((ext_vector_type(8)))  float  f32x8;

union Frag16 { bf16x16 v; unsigned u[8]; u16 h[16]; };

DEV u16 f2bf(float f) {
  unsigned u = __float_as_uint(f);
  u += 0x7FFFu + ((u >> 16) & 1u);              // round-to-nearest-even
  return (u16)(u >> 16);
}
DEV unsigned pack2bf(float lo, float hi) {
  return (unsigned)f2bf(lo) | ((unsigned)f2bf(hi) << 16);
}
DEV f32x8 f8zero() { f32x8 z = {0.f,0.f,0.f,0.f,0.f,0.f,0.f,0.f}; return z; }

DEV f32x8 wmma_bf16(const Frag16& a, const Frag16& b, f32x8 c) {
  return __builtin_amdgcn_wmma_f32_16x16x32_bf16(false, a.v, false, b.v,
                                                 (short)0, c, false, false);
}

// A-fragment (16x32 bf16) from a bf16 global row (row ptr + kbase), 2x16B.
// K(j) = 16*lanehi + j: one contiguous 16-half run.
DEV void load_afrag_h(const u16* base, int lh, Frag16& f) {
  uint4 a = *reinterpret_cast<const uint4*>(base + lh * 16);
  uint4 b = *reinterpret_cast<const uint4*>(base + lh * 16 + 8);
  f.u[0] = a.x; f.u[1] = a.y; f.u[2] = a.z; f.u[3] = a.w;
  f.u[4] = b.x; f.u[5] = b.y; f.u[6] = b.z; f.u[7] = b.w;
}
// B-fragment (32x16 bf16): N fixed per lane, K = 16*lanehi + j, one 32B run.
DEV void load_bfrag_h(const u16* base, int lh, Frag16& f) {
  uint4 a = *reinterpret_cast<const uint4*>(base + lh * 16);
  uint4 b = *reinterpret_cast<const uint4*>(base + lh * 16 + 8);
  f.u[0] = a.x; f.u[1] = a.y; f.u[2] = a.z; f.u[3] = a.w;
  f.u[4] = b.x; f.u[5] = b.y; f.u[6] = b.z; f.u[7] = b.w;
}

// ---------------------------------------------------------------------------
// Tiled bf16 WMMA GEMM, bf16 in / f32 accumulate, f32 and/or bf16 out:
//   BT=false: B is [K,N] bf16 (ldb=N).   BT=true: B is [N,K] bf16 (ldb=K).
// Block tile 64x64, 8 waves (4m x 2n), two 16x16 accumulators per wave.
// All M,N divisible by 64 and K divisible by 32 in this workload.
// ---------------------------------------------------------------------------
template <bool BT>
__global__ void __launch_bounds__(256)
gemm_bf16_kernel(const u16* __restrict__ A, const u16* __restrict__ Bm,
                 float* __restrict__ C, u16* __restrict__ Ch,
                 const float* __restrict__ Add,
                 int M, int N, int K,
                 long long sA, long long sB, long long sC, float alpha)
{
  A  += (long long)blockIdx.z * sA;
  Bm += (long long)blockIdx.z * sB;
  if (C)   C   += (long long)blockIdx.z * sC;
  if (Ch)  Ch  += (long long)blockIdx.z * sC;
  if (Add) Add += (long long)blockIdx.z * sC;

  const int m0 = blockIdx.y * 64, n0 = blockIdx.x * 64;
  const int t = threadIdx.x;
  __shared__ __align__(16) u16 As[64 * 34];   // [m][k] bf16
  __shared__ __align__(16) u16 Bs[64 * 34];   // [n][k] bf16
  unsigned* As32w = (unsigned*)As;
  unsigned* Bs32w = (unsigned*)Bs;
  const unsigned* As32 = (const unsigned*)As;
  const unsigned* Bs32 = (const unsigned*)Bs;
  const unsigned* Ag32 = (const unsigned*)A;
  const unsigned* Bg32 = (const unsigned*)Bm;

  const int wave = t >> 5, lane = t & 31;
  const int wm = wave & 3, wn = wave >> 2;
  const int lh = lane >> 4, ln = lane & 15;
  const int Kw = K >> 1;                       // row pitch in u32

  f32x8 acc0 = f8zero(), acc1 = f8zero();

  for (int k0 = 0; k0 < K; k0 += 32) {
    const int kp = k0 >> 1;
#pragma unroll
    for (int i = 0; i < 4; ++i) {              // A tile 64x32: raw u32 copy
      int e = i * 256 + t, r = e >> 4, cp = e & 15;
      As32w[r * 17 + cp] = Ag32[(size_t)(m0 + r) * Kw + kp + cp];
    }
    if (BT) {
#pragma unroll
      for (int i = 0; i < 4; ++i) {            // B tile: raw u32 copy
        int e = i * 256 + t, n = e >> 4, cp = e & 15;
        Bs32w[n * 17 + cp] = Bg32[(size_t)(n0 + n) * Kw + kp + cp];
      }
    } else {
#pragma unroll
      for (int i = 0; i < 8; ++i) {            // B tile [K,N] -> [n][k]
        int e = i * 256 + t, k = e >> 6, n = e & 63;
        Bs[n * 34 + k] = Bm[(size_t)(k0 + k) * N + n0 + n];
      }
    }
    if (k0 + 32 < K)
      __builtin_prefetch(&A[(size_t)(m0 + (t & 63)) * K + k0 + 32], 0, 1);
    __syncthreads();

    Frag16 af, bf0, bf1;
    const int arow = wm * 16 + ln;
#pragma unroll
    for (int p = 0; p < 4; ++p) {
      af.u[p]     = As32[arow * 17 + lh * 8 + p];
      af.u[4 + p] = As32[arow * 17 + lh * 8 + 4 + p];
    }
    const int nr0 = wn * 32 + ln;
#pragma unroll
    for (int q = 0; q < 8; ++q) {
      bf0.u[q] = Bs32[nr0 * 17 + lh * 8 + q];
      bf1.u[q] = Bs32[(nr0 + 16) * 17 + lh * 8 + q];
    }
    acc0 = wmma_bf16(af, bf0, acc0);
    acc1 = wmma_bf16(af, bf1, acc1);
    __syncthreads();
  }

#pragma unroll
  for (int r = 0; r < 8; ++r) {
    int row = m0 + wm * 16 + r + lh * 8;
    int col = n0 + wn * 32 + ln;
    if (row < M && col + 16 < N) {
      size_t i0 = (size_t)row * N + col, i1 = i0 + 16;
      float v0 = acc0[r] * alpha, v1 = acc1[r] * alpha;
      if (Add) { v0 += Add[i0]; v1 += Add[i1]; }
      if (C)  { C[i0] = v0;       C[i1] = v1; }
      if (Ch) { Ch[i0] = f2bf(v0); Ch[i1] = f2bf(v1); }
    }
  }
}

// ---------------------------------------------------------------------------
// Flash attention, one wave per (b, h, 16-query tile). Causal.
// Q/K/V are bf16; QK^T and P.V both via v_wmma_f32_16x16x32_bf16.
// Output written bf16 (feeds the WO GEMM).
// ---------------------------------------------------------------------------
__global__ void __launch_bounds__(32)
attn_kernel(const u16* __restrict__ Q, const u16* __restrict__ Kg,
            const u16* __restrict__ Vg, u16* __restrict__ O)
{
  const int qt = blockIdx.x, h = blockIdx.y, b = blockIdx.z;
  const int lane = threadIdx.x, lh = lane >> 4, ln = lane & 15;
  const int q0 = qt * 16;

  __shared__ __align__(16) u16 Pls[16 * 34];  // P tile bf16 [row][col]
  __shared__ __align__(16) u16 Vls[64 * 34];  // V^T tile bf16 [dh][key]
  const unsigned* P32 = (const unsigned*)Pls;
  const unsigned* V32 = (const unsigned*)Vls;

  Frag16 qf0, qf1;
  const size_t qbase = ((size_t)(b * SS + q0 + ln)) * DD + h * DH;
  load_afrag_h(Q + qbase,      lh, qf0);   // dh 0..31
  load_afrag_h(Q + qbase + 32, lh, qf1);   // dh 32..63

  f32x8 o0 = f8zero(), o1 = f8zero(), o2 = f8zero(), o3 = f8zero();
  float mi[8], li[8];
#pragma unroll
  for (int r = 0; r < 8; ++r) { mi[r] = -3.0e38f; li[r] = 0.f; }

  const int kbmax = (q0 + 15) >> 5;
  for (int kb = 0; kb <= kbmax; ++kb) {
    // ---- scores S = Q . K^T (bf16 K rows are contiguous per lane) ----
    Frag16 kf00, kf01, kf10, kf11;
    {
      int key0 = kb * 32 + ln, key1 = key0 + 16;
      const u16* kr0 = Kg + ((size_t)(b * SS + key0)) * DD + h * DH;
      const u16* kr1 = Kg + ((size_t)(b * SS + key1)) * DD + h * DH;
      load_bfrag_h(kr0,      lh, kf00); load_bfrag_h(kr0 + 32, lh, kf01);
      load_bfrag_h(kr1,      lh, kf10); load_bfrag_h(kr1 + 32, lh, kf11);
    }
    f32x8 s0 = f8zero(), s1 = f8zero();
    s0 = wmma_bf16(qf0, kf00, s0); s0 = wmma_bf16(qf1, kf01, s0);
    s1 = wmma_bf16(qf0, kf10, s1); s1 = wmma_bf16(qf1, kf11, s1);

    // ---- stage V tile transposed into LDS ([dh][key], raw bf16 copy) ----
#pragma unroll
    for (int it = 0; it < 8; ++it) {
      int idx8 = (it * 32 + lane) * 8;        // 32 keys x 64 dh halves
      int key = idx8 >> 6, dh = idx8 & 63;
      uint4 w = *reinterpret_cast<const uint4*>(
          Vg + ((size_t)(b * SS + kb * 32 + key)) * DD + h * DH + dh);
      Vls[(dh + 0) * 34 + key] = (u16)(w.x);
      Vls[(dh + 1) * 34 + key] = (u16)(w.x >> 16);
      Vls[(dh + 2) * 34 + key] = (u16)(w.y);
      Vls[(dh + 3) * 34 + key] = (u16)(w.y >> 16);
      Vls[(dh + 4) * 34 + key] = (u16)(w.z);
      Vls[(dh + 5) * 34 + key] = (u16)(w.z >> 16);
      Vls[(dh + 6) * 34 + key] = (u16)(w.w);
      Vls[(dh + 7) * 34 + key] = (u16)(w.w >> 16);
    }

    // ---- causal mask + online softmax (row reductions across 16 lanes) ----
#pragma unroll
    for (int r = 0; r < 8; ++r) {
      int qi = q0 + r + lh * 8;
      int c0 = kb * 32 + ln, c1 = c0 + 16;
      float v0 = (c0 <= qi) ? s0[r] * 0.125f : -3.0e38f;   // 1/sqrt(64)
      float v1 = (c1 <= qi) ? s1[r] * 0.125f : -3.0e38f;
      float mx = fmaxf(v0, v1);
      mx = fmaxf(mx, __shfl_xor(mx, 1));
      mx = fmaxf(mx, __shfl_xor(mx, 2));
      mx = fmaxf(mx, __shfl_xor(mx, 4));
      mx = fmaxf(mx, __shfl_xor(mx, 8));
      float mnew = fmaxf(mi[r], mx);
      float p0 = __expf(v0 - mnew), p1 = __expf(v1 - mnew);
      float rs = p0 + p1;
      rs += __shfl_xor(rs, 1); rs += __shfl_xor(rs, 2);
      rs += __shfl_xor(rs, 4); rs += __shfl_xor(rs, 8);
      float resc = __expf(mi[r] - mnew);
      li[r] = li[r] * resc + rs;
      mi[r] = mnew;
      o0[r] *= resc; o1[r] *= resc; o2[r] *= resc; o3[r] *= resc;
      Pls[(r + lh * 8) * 34 + ln]      = f2bf(p0);
      Pls[(r + lh * 8) * 34 + 16 + ln] = f2bf(p1);
    }
    __syncthreads();

    // ---- O += P (A-frag via LDS) . V (B-frag from transposed LDS tile) ----
    Frag16 pf;
#pragma unroll
    for (int p = 0; p < 4; ++p) {
      pf.u[p]     = P32[ln * 17 + lh * 8 + p];
      pf.u[4 + p] = P32[ln * 17 + lh * 8 + 4 + p];
    }
#pragma unroll
    for (int nt = 0; nt < 4; ++nt) {
      Frag16 vf;
      int n = nt * 16 + ln;
#pragma unroll
      for (int q = 0; q < 8; ++q) vf.u[q] = V32[n * 17 + lh * 8 + q];
      if      (nt == 0) o0 = wmma_bf16(pf, vf, o0);
      else if (nt == 1) o1 = wmma_bf16(pf, vf, o1);
      else if (nt == 2) o2 = wmma_bf16(pf, vf, o2);
      else              o3 = wmma_bf16(pf, vf, o3);
    }
    __syncthreads();
  }

#pragma unroll
  for (int r = 0; r < 8; ++r) {
    float inv = 1.0f / li[r];
    size_t base = ((size_t)(b * SS + q0 + r + lh * 8)) * DD + h * DH + ln;
    O[base]      = f2bf(o0[r] * inv);
    O[base + 16] = f2bf(o1[r] * inv);
    O[base + 32] = f2bf(o2[r] * inv);
    O[base + 48] = f2bf(o3[r] * inv);
  }
}

// ---------------------------------------------------------------------------
// One-shot f32 -> packed bf16 convert (for WO and knowledge_K).
// ---------------------------------------------------------------------------
__global__ void __launch_bounds__(256)
cvt_bf16_kernel(const float* __restrict__ in, u16* __restrict__ out, int n2)
{
  int i = blockIdx.x * 256 + threadIdx.x;
  if (i < n2)
    ((unsigned*)out)[i] = pack2bf(in[2 * i], in[2 * i + 1]);
}

// ---------------------------------------------------------------------------
// LayerNorm over D=1024, one block per token; writes f32 (routers) + bf16
// (WMMA consumers) in a single pass.
// ---------------------------------------------------------------------------
__global__ void __launch_bounds__(256)
ln_kernel(const float* __restrict__ x, const float* __restrict__ g,
          const float* __restrict__ bta, float* __restrict__ nx,
          u16* __restrict__ nxh)
{
  const int tok = blockIdx.x, t = threadIdx.x;
  __shared__ float row[DD];
  __shared__ float rs[8], rs2[8];
  float s = 0.f, s2 = 0.f;
  for (int i = t; i < DD; i += 256) {
    float v = x[(size_t)tok * DD + i];
    row[i] = v; s += v; s2 += v * v;
  }
  s  += __shfl_xor(s, 16);  s  += __shfl_xor(s, 8);  s  += __shfl_xor(s, 4);
  s  += __shfl_xor(s, 2);   s  += __shfl_xor(s, 1);
  s2 += __shfl_xor(s2, 16); s2 += __shfl_xor(s2, 8); s2 += __shfl_xor(s2, 4);
  s2 += __shfl_xor(s2, 2);  s2 += __shfl_xor(s2, 1);
  const int wave = t >> 5, lane = t & 31;
  if (lane == 0) { rs[wave] = s; rs2[wave] = s2; }
  __syncthreads();
  if (t == 0) {
    float a = 0.f, b = 0.f;
    for (int w = 0; w < 8; ++w) { a += rs[w]; b += rs2[w]; }
    float mean = a * (1.0f / DD);
    float var = b * (1.0f / DD) - mean * mean;
    rs[0] = mean; rs2[0] = rsqrtf(var + 1e-5f);
  }
  __syncthreads();
  const float mean = rs[0], inv = rs2[0];
  for (int i = 2 * t; i < DD; i += 512) {
    float a = (row[i]     - mean) * inv * g[i]     + bta[i];
    float b = (row[i + 1] - mean) * inv * g[i + 1] + bta[i + 1];
    nx[(size_t)tok * DD + i]     = a;
    nx[(size_t)tok * DD + i + 1] = b;
    ((unsigned*)nxh)[((size_t)tok * DD + i) >> 1] = pack2bf(a, b);
  }
}

// ---------------------------------------------------------------------------
// Router: per-token logits (64 experts), softmax, weight by importance.
// pref[tok][e] = softmax_e(nx_tok . W_e) * importance[tok]   (no atomics)
// ---------------------------------------------------------------------------
__global__ void __launch_bounds__(256)
router_kernel(const float* __restrict__ nx, const float* __restrict__ W,
              const float* __restrict__ imp, float* __restrict__ pref)
{
  const int tok = blockIdx.x, t = threadIdx.x;
  __shared__ float row[DD];
  __shared__ float logits[NEXP];
  __shared__ float red2[2];
  for (int i = t; i < DD; i += 256) row[i] = nx[(size_t)tok * DD + i];
  __syncthreads();
  const int wave = t >> 5, lane = t & 31;
  for (int e8 = 0; e8 < 8; ++e8) {
    int e = wave * 8 + e8;
    float acc = 0.f;
    for (int d = lane; d < DD; d += 32) acc += row[d] * W[(size_t)e * DD + d];
    acc += __shfl_xor(acc, 16); acc += __shfl_xor(acc, 8);
    acc += __shfl_xor(acc, 4);  acc += __shfl_xor(acc, 2);
    acc += __shfl_xor(acc, 1);
    if (lane == 0) logits[e] = acc;
  }
  __syncthreads();
  if (t == 0) {
    float mx = -3.0e38f;
    for (int i = 0; i < NEXP; ++i) mx = fmaxf(mx, logits[i]);
    float sum = 0.f;
    for (int i = 0; i < NEXP; ++i) sum += __expf(logits[i] - mx);
    red2[0] = mx; red2[1] = sum;
  }
  __syncthreads();
  if (t < NEXP) {
    float p = __expf(logits[t] - red2[0]) / red2[1];
    pref[(size_t)tok * NEXP + t] = p * imp[tok];
  }
}

// Deterministic pooled reduction: pool[b][n] = sum_s pref[b][s][n]
__global__ void __launch_bounds__(256)
pool_reduce_kernel(const float* __restrict__ pref, float* __restrict__ pool)
{
  const int n = blockIdx.x, b = blockIdx.y, t = threadIdx.x;
  float s = 0.f;
  for (int sdx = t; sdx < SS; sdx += 256)
    s += pref[((size_t)(b * SS + sdx)) * NEXP + n];
  s += __shfl_xor(s, 16); s += __shfl_xor(s, 8); s += __shfl_xor(s, 4);
  s += __shfl_xor(s, 2);  s += __shfl_xor(s, 1);
  __shared__ float red[8];
  const int wave = t >> 5, lane = t & 31;
  if (lane == 0) red[wave] = s;
  __syncthreads();
  if (t == 0) {
    float a = 0.f;
    for (int w = 0; w < 8; ++w) a += red[w];
    pool[b * NEXP + n] = a;
  }
}

__global__ void __launch_bounds__(64)
normpool_kernel(float* __restrict__ pool)
{
  const int b = blockIdx.x, t = threadIdx.x;
  __shared__ float v[NEXP];
  __shared__ float ssum;
  v[t] = pool[b * NEXP + t];
  __syncthreads();
  if (t == 0) {
    float s = 0.f;
    for (int i = 0; i < NEXP; ++i) s += v[i];
    ssum = s + 1e-8f;
  }
  __syncthreads();
  pool[b * NEXP + t] = v[t] / ssum;
}

// ---------------------------------------------------------------------------
// Expert-pool mixes: read the 33MB pool exactly once, accumulate all batches,
// write bf16 (consumed only by WMMA GEMMs).
// ---------------------------------------------------------------------------
__global__ void __launch_bounds__(256)
mix_compress_kernel(const float* __restrict__ CN, const float* __restrict__ w,
                    u16* __restrict__ out)
{
  const size_t e = (size_t)blockIdx.x * 256 + threadIdx.x;   // < 131072
  __shared__ float ws[BB * NEXP];
  ws[threadIdx.x] = w[threadIdx.x];
  __syncthreads();
  float a0 = 0.f, a1 = 0.f, a2 = 0.f, a3 = 0.f;
  for (int n = 0; n < NEXP; ++n) {
    float c = CN[(size_t)n * (DD * RR) + e];
    a0 += ws[n] * c;       a1 += ws[64 + n] * c;
    a2 += ws[128 + n] * c; a3 += ws[192 + n] * c;
  }
  out[e]                          = f2bf(a0);
  out[(size_t)(DD * RR) + e]      = f2bf(a1);
  out[(size_t)2 * (DD * RR) + e]  = f2bf(a2);
  out[(size_t)3 * (DD * RR) + e]  = f2bf(a3);
}

__global__ void __launch_bounds__(256)
mix_expand_kernel(const float* __restrict__ EP,
                  const float* __restrict__ wq, const float* __restrict__ wk,
                  const float* __restrict__ wv,
                  u16* __restrict__ eQ, u16* __restrict__ eK,
                  u16* __restrict__ eV)
{
  const size_t e = (size_t)blockIdx.x * 256 + threadIdx.x;   // < 131072
  __shared__ float ws[3 * BB * NEXP];
  {
    int t = threadIdx.x;
    ws[t]       = wq[t];
    ws[256 + t] = wk[t];
    ws[512 + t] = wv[t];
  }
  __syncthreads();
  float aq[BB] = {0.f,0.f,0.f,0.f}, ak[BB] = {0.f,0.f,0.f,0.f},
        av[BB] = {0.f,0.f,0.f,0.f};
  for (int n = 0; n < NEXP; ++n) {
    float c = EP[(size_t)n * (RR * DD) + e];
#pragma unroll
    for (int b = 0; b < BB; ++b) {
      aq[b] += ws[b * 64 + n] * c;
      ak[b] += ws[256 + b * 64 + n] * c;
      av[b] += ws[512 + b * 64 + n] * c;
    }
  }
#pragma unroll
  for (int b = 0; b < BB; ++b) {
    eQ[(size_t)b * (RR * DD) + e] = f2bf(aq[b]);
    eK[(size_t)b * (RR * DD) + e] = f2bf(ak[b]);
    eV[(size_t)b * (RR * DD) + e] = f2bf(av[b]);
  }
}

// ---------------------------------------------------------------------------
// Top-8 over 4096 knowledge scores per token, softmax, gather-weighted sum of
// knowledge_V, fused with final residual (in-place on d_out).
// ---------------------------------------------------------------------------
__global__ void __launch_bounds__(256)
topk_kernel(const float* __restrict__ ks, const float* __restrict__ kV,
            float* __restrict__ out)
{
  const int tok = blockIdx.x, t = threadIdx.x;
  __shared__ float sc_sh[NK];
  __shared__ float redv[256];
  __shared__ int   redi[256];
  __shared__ float selw[TOPK];
  __shared__ int   seli[TOPK];
  const float scale = 0.08838834764831845f;   // 1/sqrt(128)

  for (int i = t; i < NK; i += 256)
    sc_sh[i] = ks[(size_t)tok * NK + i] * scale;
  __syncthreads();

  for (int kk = 0; kk < TOPK; ++kk) {
    float bv = -3.0e38f; int bi = NK;
    for (int i = t; i < NK; i += 256) {
      float v = sc_sh[i];
      if (v > bv || (v == bv && i < bi)) { bv = v; bi = i; }
    }
    redv[t] = bv; redi[t] = bi;
    __syncthreads();
    for (int off = 128; off > 0; off >>= 1) {
      if (t < off) {
        float ov = redv[t + off]; int oi = redi[t + off];
        if (ov > redv[t] || (ov == redv[t] && oi < redi[t])) {
          redv[t] = ov; redi[t] = oi;
        }
      }
      __syncthreads();
    }
    if (t == 0) {
      selw[kk] = redv[0]; seli[kk] = redi[0];
      sc_sh[redi[0]] = -3.0e38f;
    }
    __syncthreads();
  }
  if (t == 0) {   // softmax over the 8 (selw[0] is the max)
    float mx = selw[0], s = 0.f, e[TOPK];
    for (int k = 0; k < TOPK; ++k) { e[k] = __expf(selw[k] - mx); s += e[k]; }
    for (int k = 0; k < TOPK; ++k) selw[k] = e[k] / s;
  }
  __syncthreads();
  for (int d = t; d < DD; d += 256) {
    float acc = out[(size_t)tok * DD + d];
#pragma unroll
    for (int k = 0; k < TOPK; ++k)
      acc += selw[k] * kV[(size_t)seli[k] * DD + d];
    out[(size_t)tok * DD + d] = acc;
  }
}

// ---------------------------------------------------------------------------
// Host-side orchestration
// ---------------------------------------------------------------------------
extern "C" void kernel_launch(void* const* d_in, const int* in_sizes, int n_in,
                              void* d_out, int out_size, void* d_ws,
                              size_t ws_size, hipStream_t stream)
{
  (void)in_sizes; (void)n_in; (void)out_size; (void)ws_size;
  const float* x   = (const float*)d_in[0];
  const float* imp = (const float*)d_in[1];
  // d_in[2] = mask (causal structure is hard-coded)
  const float* CN  = (const float*)d_in[3];
  const float* EP  = (const float*)d_in[4];
  const float* kK  = (const float*)d_in[5];
  const float* kV  = (const float*)d_in[6];
  const float* Wc  = (const float*)d_in[7];
  const float* WQr = (const float*)d_in[8];
  const float* WKr = (const float*)d_in[9];
  const float* WVr = (const float*)d_in[10];
  const float* Wm  = (const float*)d_in[11];
  const float* WO  = (const float*)d_in[12];
  const float* g1  = (const float*)d_in[13];
  const float* b1  = (const float*)d_in[14];
  const float* g2  = (const float*)d_in[15];
  const float* b2  = (const float*)d_in[16];
  float* out = (float*)d_out;

  char* ws = (char*)d_ws;
  size_t off = 0;
  auto alloc = [&](size_t bytes) -> char* {
    char* p = ws + off;
    off += (bytes + 255) & ~(size_t)255;
    return p;
  };
  const size_t BSD = (size_t)BB * SS * DD;       // 4.19M elems
  const size_t BSR = (size_t)BB * SS * RR;       // 524K elems
  float* nx    = (float*)alloc(BSD * 4);
  u16*   nxh   = (u16*)  alloc(BSD * 2);
  u16*   aoh   = (u16*)  alloc(BSD * 2);         // attention output (bf16)
  u16*   Qh    = (u16*)  alloc(BSD * 2);
  u16*   Kh    = (u16*)  alloc(BSD * 2);
  u16*   Vh    = (u16*)  alloc(BSD * 2);
  u16*   hb    = (u16*)  alloc(BSR * 2);         // h / Qm (bf16)
  u16*   scb   = (u16*)  alloc(BSR * 2);         // sc / scm (bf16)
  u16*   eQ    = (u16*)  alloc(BSR * 2);
  u16*   eK    = (u16*)  alloc(BSR * 2);
  u16*   eV    = (u16*)  alloc(BSR * 2);
  u16*   WOh   = (u16*)  alloc((size_t)DD * DD * 2);
  u16*   kKh   = (u16*)  alloc((size_t)NK * RR * 2);
  float* ksb   = (float*)alloc((size_t)BB * SS * NK * 4);
  float* pref  = (float*)alloc((size_t)BB * SS * NEXP * 4);
  float* poolc = (float*)alloc(BB * NEXP * 4);
  float* poolq = (float*)alloc(BB * NEXP * 4);
  float* poolk = (float*)alloc(BB * NEXP * 4);
  float* poolv = (float*)alloc(BB * NEXP * 4);
  float* poolm = (float*)alloc(BB * NEXP * 4);

  const long long sBSD = (long long)SS * DD;     // per-batch stride (elems)
  const long long sBSR = (long long)SS * RR;

  // one-shot bf16 packs of the two WMMA-consumed weights
  cvt_bf16_kernel<<<(DD * DD / 2 + 255) / 256, 256, 0, stream>>>(WO, WOh,
                                                                 DD * DD / 2);
  cvt_bf16_kernel<<<(NK * RR / 2 + 255) / 256, 256, 0, stream>>>(kK, kKh,
                                                                 NK * RR / 2);

  // ---- attention sub-block ----
  ln_kernel<<<BB * SS, 256, 0, stream>>>(x, g1, b1, nx, nxh);

  const float* rW[4] = {Wc, WQr, WKr, WVr};
  float*       rP[4] = {poolc, poolq, poolk, poolv};
  for (int i = 0; i < 4; ++i) {
    router_kernel<<<BB * SS, 256, 0, stream>>>(nx, rW[i], imp, pref);
    pool_reduce_kernel<<<dim3(NEXP, BB), 256, 0, stream>>>(pref, rP[i]);
    normpool_kernel<<<BB, 64, 0, stream>>>(rP[i]);
  }
  mix_compress_kernel<<<512, 256, 0, stream>>>(CN, poolc, scb);
  mix_expand_kernel<<<512, 256, 0, stream>>>(EP, poolq, poolk, poolv,
                                             eQ, eK, eV);

  // h = nx @ sc          [S=1024, R=128] per batch, K=D=1024  (bf16 out)
  gemm_bf16_kernel<false><<<dim3(RR / 64, SS / 64, BB), 256, 0, stream>>>(
      nxh, scb, nullptr, hb, nullptr, SS, RR, DD, sBSD, sBSR, sBSR, 1.0f);
  // Q/K/V = h @ e{Q,K,V}  [S=1024, D=1024] per batch, K=R=128 (bf16 out)
  gemm_bf16_kernel<false><<<dim3(DD / 64, SS / 64, BB), 256, 0, stream>>>(
      hb, eQ, nullptr, Qh, nullptr, SS, DD, RR, sBSR, sBSR, sBSD, 1.0f);
  gemm_bf16_kernel<false><<<dim3(DD / 64, SS / 64, BB), 256, 0, stream>>>(
      hb, eK, nullptr, Kh, nullptr, SS, DD, RR, sBSR, sBSR, sBSD, 1.0f);
  gemm_bf16_kernel<false><<<dim3(DD / 64, SS / 64, BB), 256, 0, stream>>>(
      hb, eV, nullptr, Vh, nullptr, SS, DD, RR, sBSR, sBSR, sBSD, 1.0f);

  attn_kernel<<<dim3(SS / 16, HH, BB), 32, 0, stream>>>(Qh, Kh, Vh, aoh);

  // x1 = attn_out @ WO^T + x   -> d_out   (M=B*S=4096, N=K=1024)
  gemm_bf16_kernel<true><<<dim3(DD / 64, (BB * SS) / 64, 1), 256, 0, stream>>>(
      aoh, WOh, out, nullptr, x, BB * SS, DD, DD, 0, 0, 0, 1.0f);

  // ---- memory sub-block ----
  ln_kernel<<<BB * SS, 256, 0, stream>>>(out, g2, b2, nx, nxh);
  router_kernel<<<BB * SS, 256, 0, stream>>>(nx, Wm, imp, pref);
  pool_reduce_kernel<<<dim3(NEXP, BB), 256, 0, stream>>>(pref, poolm);
  normpool_kernel<<<BB, 64, 0, stream>>>(poolm);
  mix_compress_kernel<<<512, 256, 0, stream>>>(CN, poolm, scb);

  // Qm = nx2 @ scm       [S, R] per batch (bf16 out)
  gemm_bf16_kernel<false><<<dim3(RR / 64, SS / 64, BB), 256, 0, stream>>>(
      nxh, scb, nullptr, hb, nullptr, SS, RR, DD, sBSD, sBSR, sBSR, 1.0f);
  // kscores = Qm @ knowledge_K^T   (M=4096, N=4096, K=128, f32 out)
  gemm_bf16_kernel<true><<<dim3(NK / 64, (BB * SS) / 64, 1), 256, 0, stream>>>(
      hb, kKh, ksb, nullptr, nullptr, BB * SS, NK, RR, 0, 0, 0, 1.0f);

  // top-8 softmax gather + final residual (in place on d_out)
  topk_kernel<<<BB * SS, 256, 0, stream>>>(ksb, kV, out);
}
